// QKGainAttention_2078764171284
// MI455X (gfx1250) — matrix operand order repaired and checked by reading, not compiled, not run on hardware
//
#include <hip/hip_runtime.h>
#include <stdint.h>

// ---------------------------------------------------------------------------
// Problem constants (from reference): B=2, T=2048, C=2048, NH=16, HD=128
// Roofline: ~206 GFLOP vs ~0.2 GB unique traffic -> strongly compute bound;
// int6 weights are exact in f16, so V_WMMA_F32_16X16X32_F16 everywhere.
// ---------------------------------------------------------------------------
#define B_SZ  2
#define T_SZ  2048
#define C_SZ  2048
#define NH_SZ 16
#define HD_SZ 128
#define M_SZ  (B_SZ * T_SZ)      // 4096 rows

typedef _Float16 f16;
typedef __attribute__((ext_vector_type(16))) _Float16 v16h;
typedef __attribute__((ext_vector_type(8)))  _Float16 v8h;
typedef __attribute__((ext_vector_type(8)))  float    v8f;
typedef __attribute__((ext_vector_type(4)))  unsigned int u32x4;
typedef __attribute__((ext_vector_type(8)))  unsigned int u32x8;

// Build a 16-half A/B fragment from two 16-byte chunks (CDNA5 16-bit WMMA
// VGPR striping: two contiguous 8-half groups per lane).
static __device__ inline v16h make_v16(const f16* lo, const f16* hi) {
  v8h a = *(const v8h*)lo;
  v8h b = *(const v8h*)hi;
  v16h r;
#pragma unroll
  for (int i = 0; i < 8; ++i) { r[i] = a[i]; r[i + 8] = b[i]; }
  return r;
}

static __device__ inline v8f wmma_f16(v16h a, v16h b, v8f c) {
  return __builtin_amdgcn_wmma_f32_16x16x32_f16(false, a, false, b,
                                                (short)0, c, false, false);
}

static __device__ inline void st_out(float* p, float v) { *p = v; }
static __device__ inline void st_out(f16* p,   float v) { *p = (f16)v; }

// Load 32 elements of A into 4 v8h registers (f32 converts on the fly).
static __device__ inline void ldgA(const float* p, v8h* r) {
  const float4* s = (const float4*)p;
#pragma unroll
  for (int i = 0; i < 4; ++i) {
    float4 a = s[2 * i], b = s[2 * i + 1];
    v8h h;
    h[0] = (f16)a.x; h[1] = (f16)a.y; h[2] = (f16)a.z; h[3] = (f16)a.w;
    h[4] = (f16)b.x; h[5] = (f16)b.y; h[6] = (f16)b.z; h[7] = (f16)b.w;
    r[i] = h;
  }
}
static __device__ inline void ldgA(const f16* p, v8h* r) {
#pragma unroll
  for (int i = 0; i < 4; ++i) r[i] = ((const v8h*)p)[i];
}

// Generic -> LDS byte offset (for the TDM descriptor's lds_addr field).
static __device__ inline unsigned lds_addr_of(const void* p) {
  return (unsigned)(uintptr_t)(__attribute__((address_space(3))) const void*)p;
}

// ---------------------------------------------------------------------------
// TDM: async DMA of one 32x128 f16 tile (rows of K) global -> LDS, with
// per-row padding of 4 DWORDs every 64 DWORDs (=> 136-half LDS row stride,
// conflict-free column-parallel b128 reads). 2D D# per cdna5 ISA ch.8.
// ---------------------------------------------------------------------------
static __device__ inline void tdm_load_tile_k(const f16* gsrc, unsigned lds_byte) {
  const unsigned long long ga = (unsigned long long)(uintptr_t)gsrc;
  u32x4 g0;
  g0[0] = 1u;                                  // count=1, load, no gather
  g0[1] = lds_byte;                            // lds_addr
  g0[2] = (unsigned)ga;                        // global_addr[31:0]
  g0[3] = (unsigned)(ga >> 32) | (2u << 30);   // global_addr[56:32] | type=2
  u32x8 g1;
  g1[0] = (1u << 16)                           // data_size = 2 bytes
        | (1u << 20)                           // pad_enable
        | (5u << 22)                           // pad_interval: 64 DWORDs
        | (3u << 25);                          // pad_amount: 4 DWORDs
  g1[1] = (HD_SZ & 0xFFFFu) << 16;             // tensor_dim0 = 128 (lo16)
  g1[2] = (T_SZ & 0xFFFFu) << 16;              // dim0 hi | tensor_dim1 = 2048
  g1[3] = (HD_SZ << 16);                       // dim1 hi | tile_dim0 = 128
  g1[4] = 32u;                                 // tile_dim1 = 32, tile_dim2 = 0
  g1[5] = HD_SZ;                               // tensor_dim0_stride = 128
  g1[6] = 0u;
  g1[7] = 0u;
  asm volatile("tensor_load_to_lds %0, %1" :: "s"(g0), "s"(g1) : "memory");
}

// ---------------------------------------------------------------------------
// GEMM: D[m][n] = (sum_k A[m][k] * (f16)W_int8[n][k]) * scales[n]
// Block tile 128x256, BK=64, LDS double buffered. 8 waves as 2(M)x4(N),
// each computing a 64x64 subtile: 4x4 accumulators, 32 WMMAs per K-stage.
// ---------------------------------------------------------------------------
template <typename AT, typename OT>
__global__ __launch_bounds__(256)
void gemm_w6(const AT* __restrict__ A, const signed char* __restrict__ W,
             const float* __restrict__ scales, OT* __restrict__ D,
             int M, int N, int K)
{
  __shared__ alignas(16) f16 As[2][128][72];   // [m][k] + 8 pad halves
  __shared__ alignas(16) f16 Ws[2][256][72];   // [n][k] + 8 pad halves

  const int m0   = blockIdx.y * 128;
  const int n0   = blockIdx.x * 256;
  const int tid  = threadIdx.x;
  const int lane = tid & 31;
  const int wv   = tid >> 5;     // wave 0..7
  const int wm   = wv & 1;       // 2 wave tiles along M (64 rows)
  const int wn   = wv >> 1;      // 4 wave tiles along N (64 cols)
  const int lm   = lane & 15;
  const int lh   = lane >> 4;

  const int frow = tid >> 1;     // 0..127 (A fill row)
  const int fseg = tid & 1;      // 0..1   (32-element segment)

  v8f acc[4][4];
#pragma unroll
  for (int i = 0; i < 4; ++i)
#pragma unroll
    for (int j = 0; j < 4; ++j)
#pragma unroll
      for (int e = 0; e < 8; ++e) acc[i][j][e] = 0.0f;

  v8h ar[4];     // 32 A elements (f16)
  int4 wr[4];    // 64 W bytes

  // --- global loads for stage k0 into registers ---
  auto ldg = [&](int k0) {
    const AT* ap = A + (size_t)(m0 + frow) * K + k0 + fseg * 32;
    ldgA(ap, ar);
    const signed char* wp = W + (size_t)(n0 + tid) * K + k0;
#pragma unroll
    for (int i = 0; i < 4; ++i) wr[i] = ((const int4*)wp)[i];
    if (k0 + 64 < K) {
      __builtin_prefetch(ap + 64, 0, 1);       // global_prefetch_b8
      __builtin_prefetch(wp + 64, 0, 1);
    }
  };
  // --- registers -> LDS (int6 payload dequantized to exact f16) ---
  auto sts = [&](int buf) {
#pragma unroll
    for (int i = 0; i < 4; ++i) *(v8h*)&As[buf][frow][fseg * 32 + i * 8] = ar[i];
#pragma unroll
    for (int i = 0; i < 4; ++i) {
      const signed char* wb = (const signed char*)&wr[i];
      v8h h0, h1;
#pragma unroll
      for (int j = 0; j < 8; ++j) {
        h0[j] = (f16)(float)wb[j];
        h1[j] = (f16)(float)wb[8 + j];
      }
      *(v8h*)&Ws[buf][tid][i * 16]     = h0;
      *(v8h*)&Ws[buf][tid][i * 16 + 8] = h1;
    }
  };

  ldg(0);
  sts(0);
  __syncthreads();

  const int nk = K >> 6;
  for (int ki = 0; ki < nk; ++ki) {
    const int buf = ki & 1;
    if (ki + 1 < nk) ldg((ki + 1) << 6);       // overlap with WMMA burst

    // --- 2 K-fragments x 16 WMMAs (all frags loaded before the burst) ---
#pragma unroll
    for (int kk = 0; kk < 2; ++kk) {
      v16h af[4], bf[4];
#pragma unroll
      for (int mi = 0; mi < 4; ++mi) {
        const f16* r = &As[buf][wm * 64 + mi * 16 + lm][kk * 32];
        af[mi] = make_v16(r + lh * 8, r + 16 + lh * 8);
      }
#pragma unroll
      for (int ni = 0; ni < 4; ++ni) {
        const f16* r = &Ws[buf][wn * 64 + ni * 16 + lm][kk * 32];
        bf[ni] = make_v16(r + lh * 8, r + 16 + lh * 8);
      }
#pragma unroll
      for (int mi = 0; mi < 4; ++mi)
#pragma unroll
        for (int ni = 0; ni < 4; ++ni)
          acc[mi][ni] = wmma_f16(af[mi], bf[ni], acc[mi][ni]);
    }

    if (ki + 1 < nk) sts(buf ^ 1);
    __syncthreads();
  }

  // --- epilogue: fold per-output-row int6 scale ---
#pragma unroll
  for (int mi = 0; mi < 4; ++mi)
#pragma unroll
    for (int ni = 0; ni < 4; ++ni) {
      const int gr0 = m0 + wm * 64 + mi * 16 + lh * 8;
      const int gc  = n0 + wn * 64 + ni * 16 + lm;
      const float sc = scales[gc];
#pragma unroll
      for (int j = 0; j < 8; ++j)
        st_out(&D[(size_t)(gr0 + j) * N + gc], acc[mi][ni][j] * sc);
    }
}

// ---------------------------------------------------------------------------
// Kernel 2: per-head L2 normalize q,k (x qk_gain), repack to [B*NH][T][HD] f16
// ---------------------------------------------------------------------------
__global__ __launch_bounds__(128)
void pack_norm(const f16* __restrict__ qkv, const float* __restrict__ gain_p,
               f16* __restrict__ qh, f16* __restrict__ kh, f16* __restrict__ vh)
{
  const int d   = threadIdx.x;
  const int idx = blockIdx.x;             // (b*NH + h)*T + t
  const int t   = idx % T_SZ;
  const int bh  = idx / T_SZ;
  const int b   = bh / NH_SZ;
  const int h   = bh % NH_SZ;

  const size_t base = ((size_t)b * T_SZ + t) * (3 * C_SZ) + (size_t)h * HD_SZ;
  float qv = (float)qkv[base + d];
  float kv = (float)qkv[base + C_SZ + d];
  f16   vv = qkv[base + 2 * C_SZ + d];

  float sq = qv * qv, sk = kv * kv;
#pragma unroll
  for (int m = 16; m >= 1; m >>= 1) {
    sq += __shfl_xor(sq, m, 32);
    sk += __shfl_xor(sk, m, 32);
  }
  __shared__ float red[8];
  const int w = threadIdx.x >> 5;
  if ((threadIdx.x & 31) == 0) { red[w] = sq; red[4 + w] = sk; }
  __syncthreads();
  const float ssq = red[0] + red[1] + red[2] + red[3];
  const float ssk = red[4] + red[5] + red[6] + red[7];

  const float g  = *gain_p;
  const float iq = g / fmaxf(sqrtf(ssq), 1e-12f);
  const float ik = g / fmaxf(sqrtf(ssk), 1e-12f);

  const size_t o = ((size_t)bh * T_SZ + t) * HD_SZ + d;
  qh[o] = (f16)(qv * iq);
  kh[o] = (f16)(kv * ik);
  vh[o] = vv;
}

// ---------------------------------------------------------------------------
// Kernel 3: causal flash attention. 4 waves/block, each owns 16 q rows;
// K tiles staged by the Tensor Data Mover (async, double buffered,
// s_wait_tensorcnt), V staged transposed via registers overlapping compute.
// Per 32-key chunk: 8 WMMAs for S = QK^T, online softmax, 8 WMMAs for P@V.
// ---------------------------------------------------------------------------
__global__ __launch_bounds__(128)
void flash_attn(const f16* __restrict__ qh, const f16* __restrict__ kh,
                const f16* __restrict__ vh, f16* __restrict__ yh)
{
  __shared__ alignas(16) f16 Ks[2][32][HD_SZ + 8];  // TDM dest (padded rows)
  __shared__ alignas(16) f16 Vt[2][HD_SZ][40];      // [d][k] transposed
  __shared__ alignas(16) f16 Ps[4][16][40];         // per-wave P transpose

  const int bh   = blockIdx.y;
  const int b    = bh / NH_SZ;
  const int h    = bh % NH_SZ;
  const int tid  = threadIdx.x;
  const int lane = tid & 31;
  const int wv   = tid >> 5;          // 0..3
  const int lm   = lane & 15;
  const int lh   = lane >> 4;

  const int qblk  = blockIdx.x * 64;
  const int qbase = qblk + wv * 16;
  const int nch   = (qblk + 64) >> 5;   // causal: chunks of 32 keys
  // wave-uniform causal skip bound, forced scalar (keeps EXEC untouched)
  const int qlast = __builtin_amdgcn_readfirstlane(qbase + 15);

  const f16* Q  = qh + (size_t)bh * T_SZ * HD_SZ;
  const f16* Kp = kh + (size_t)bh * T_SZ * HD_SZ;
  const f16* Vp = vh + (size_t)bh * T_SZ * HD_SZ;

  // Q fragments (16 x 128) with softmax scale 1/sqrt(HD) folded in
  const f16 qs = (f16)0.08838834764831845f;
  v16h qf[4];
#pragma unroll
  for (int dk = 0; dk < 4; ++dk) {
    const f16* r = Q + (size_t)(qbase + lm) * HD_SZ + dk * 32;
    v16h a = make_v16(r + lh * 8, r + 16 + lh * 8);
#pragma unroll
    for (int i = 0; i < 16; ++i) a[i] = a[i] * qs;
    qf[dk] = a;
  }

  v8f yacc[8];
#pragma unroll
  for (int df = 0; df < 8; ++df)
#pragma unroll
    for (int e = 0; e < 8; ++e) yacc[df][e] = 0.0f;

  float mrow[8], lrow[8];
#pragma unroll
  for (int j = 0; j < 8; ++j) { mrow[j] = -1e30f; lrow[j] = 0.0f; }

  const int frow = tid >> 2;   // 0..31 (key row for V staging)
  const int fseg = tid & 3;    // 0..3  (32-col segment of HD)

  f16 vloc[32];
  auto ldg_v = [&](int kc) {
    const f16* vp = Vp + (size_t)(kc + frow) * HD_SZ + fseg * 32;
#pragma unroll
    for (int i = 0; i < 4; ++i) ((v8h*)vloc)[i] = ((const v8h*)vp)[i];
  };
  auto sts_v = [&](int buf) {
#pragma unroll
    for (int c = 0; c < 32; ++c) Vt[buf][fseg * 32 + c][frow] = vloc[c];
  };

  // prologue: async K tile 0 via TDM; V tile 0 manual
  if (wv == 0) tdm_load_tile_k(Kp, lds_addr_of(&Ks[0][0][0]));
  ldg_v(0);
  sts_v(0);

  for (int i = 0; i < nch; ++i) {
    const int kc  = i << 5;
    const int buf = i & 1;
    const bool pf = (i + 1 < nch);
    if (pf) {
      if (wv == 0)
        tdm_load_tile_k(Kp + (size_t)(kc + 32) * HD_SZ,
                        lds_addr_of(&Ks[buf ^ 1][0][0]));
      ldg_v(kc + 32);
    }
    if (wv == 0) {
      if (pf) __builtin_amdgcn_s_wait_tensorcnt(1);   // tile i landed
      else    __builtin_amdgcn_s_wait_tensorcnt(0);
    }
    __syncthreads();

    if (kc <= qlast) {               // scalar branch; EXEC stays all-ones
      // --- S = Q @ K^T: batch-load all 8 K fragments, then 8 WMMAs ---
      v16h kb[2][4];
#pragma unroll
      for (int cn = 0; cn < 2; ++cn)
#pragma unroll
        for (int dk = 0; dk < 4; ++dk) {
          const f16* r = &Ks[buf][cn * 16 + lm][dk * 32 + lh * 8];
          kb[cn][dk] = make_v16(r, r + 16);
        }
      v8f s[2];
#pragma unroll
      for (int cn = 0; cn < 2; ++cn) {
#pragma unroll
        for (int e = 0; e < 8; ++e) s[cn][e] = 0.0f;
#pragma unroll
        for (int dk = 0; dk < 4; ++dk)
          s[cn] = wmma_f16(qf[dk], kb[cn][dk], s[cn]);
      }
      // --- causal mask + online softmax (row = qbase + lh*8 + j) ---
      float corr[8];
#pragma unroll
      for (int j = 0; j < 8; ++j) {
        const int row = qbase + lh * 8 + j;
        float s0 = ((kc + lm)      <= row) ? s[0][j] : -1e30f;
        float s1 = ((kc + 16 + lm) <= row) ? s[1][j] : -1e30f;
        float mx = fmaxf(s0, s1);
#pragma unroll
        for (int m = 8; m >= 1; m >>= 1) mx = fmaxf(mx, __shfl_xor(mx, m, 32));
        const float mn = fmaxf(mrow[j], mx);
        corr[j] = __expf(mrow[j] - mn);
        const float p0 = __expf(s0 - mn);
        const float p1 = __expf(s1 - mn);
        s[0][j] = p0; s[1][j] = p1;
        float rs = p0 + p1;
#pragma unroll
        for (int m = 8; m >= 1; m >>= 1) rs += __shfl_xor(rs, m, 32);
        lrow[j] = lrow[j] * corr[j] + rs;
        mrow[j] = mn;
      }
#pragma unroll
      for (int df = 0; df < 8; ++df)
#pragma unroll
        for (int j = 0; j < 8; ++j) yacc[df][j] *= corr[j];

      // --- P: D-layout -> A-layout via per-wave LDS, then y += P @ V ---
#pragma unroll
      for (int cn = 0; cn < 2; ++cn)
#pragma unroll
        for (int j = 0; j < 8; ++j)
          Ps[wv][lh * 8 + j][cn * 16 + lm] = (f16)s[cn][j];

      const f16* pr = &Ps[wv][lm][0];
      v16h pa = make_v16(pr + lh * 8, pr + 16 + lh * 8);
      v16h vb[8];
#pragma unroll
      for (int df = 0; df < 8; ++df) {
        const f16* r = &Vt[buf][df * 16 + lm][lh * 8];
        vb[df] = make_v16(r, r + 16);
      }
#pragma unroll
      for (int df = 0; df < 8; ++df)
        yacc[df] = wmma_f16(pa, vb[df], yacc[df]);
    }

    if (pf) sts_v(buf ^ 1);
    __syncthreads();
  }

  // --- epilogue: divide by row sum, scatter back to [B,T,C] f16 ---
  float rcp[8];
#pragma unroll
  for (int j = 0; j < 8; ++j) rcp[j] = (lrow[j] > 0.0f) ? 1.0f / lrow[j] : 0.0f;
#pragma unroll
  for (int df = 0; df < 8; ++df) {
    const int dcol = h * HD_SZ + df * 16 + lm;
#pragma unroll
    for (int j = 0; j < 8; ++j) {
      const int trow = qbase + lh * 8 + j;
      yh[((size_t)b * T_SZ + trow) * C_SZ + dcol] = (f16)(yacc[df][j] * rcp[j]);
    }
  }
}

// ---------------------------------------------------------------------------
// Launcher. Workspace (f16 elements, ~117.4 MB):
//   qkv [M][3C] | qh [B*NH][T][HD] | kh ... | vh ... | yh [M][C]
// ---------------------------------------------------------------------------
extern "C" void kernel_launch(void* const* d_in, const int* in_sizes, int n_in,
                              void* d_out, int out_size, void* d_ws, size_t ws_size,
                              hipStream_t stream)
{
  const float*       x           = (const float*)d_in[0];
  const signed char* attn_w      = (const signed char*)d_in[1];
  const float*       attn_scales = (const float*)d_in[2];
  const signed char* proj_w      = (const signed char*)d_in[3];
  const float*       proj_scales = (const float*)d_in[4];
  const float*       qk_gain     = (const float*)d_in[5];
  float*             out         = (float*)d_out;

  f16* qkv = (f16*)d_ws;
  f16* qh  = qkv + (size_t)M_SZ * 3 * C_SZ;
  f16* kh  = qh  + (size_t)M_SZ * C_SZ;
  f16* vh  = kh  + (size_t)M_SZ * C_SZ;
  f16* yh  = vh  + (size_t)M_SZ * C_SZ;

  // 1) qkv = x @ attn_w^T (dequant): M=4096, N=6144, K=2048
  gemm_w6<float, f16><<<dim3(3 * C_SZ / 256, M_SZ / 128), 256, 0, stream>>>(
      x, attn_w, attn_scales, qkv, M_SZ, 3 * C_SZ, C_SZ);

  // 2) per-head l2norm(q,k) * qk_gain, pack head-major f16
  pack_norm<<<dim3(B_SZ * NH_SZ * T_SZ), 128, 0, stream>>>(
      qkv, qk_gain, qh, kh, vh);

  // 3) causal flash attention (TDM-staged K tiles) -> yh [B,T,C]
  flash_attn<<<dim3(T_SZ / 64, B_SZ * NH_SZ), 128, 0, stream>>>(
      qh, kh, vh, yh);

  // 4) out = yh @ proj_w^T (dequant): M=4096, N=2048, K=2048
  gemm_w6<f16, float><<<dim3(C_SZ / 256, M_SZ / 128), 256, 0, stream>>>(
      yh, proj_w, proj_scales, out, M_SZ, C_SZ, C_SZ);
}
